// q_convMixerBlock_90443421319685
// MI455X (gfx1250) — 8-bit matrix operand order repaired and checked by reading, compiled, not run on hardware
//
#include <hip/hip_runtime.h>
#include <math.h>

typedef __attribute__((ext_vector_type(8))) int v8i;

#define DIM   768
#define HW    1024
#define BATCH 16
#define TOTAL (BATCH * DIM * HW)

__device__ __forceinline__ float gelu_exact(float x) {
    return 0.5f * x * (1.0f + erff(x * 0.70710678118654752440f));
}

__device__ __forceinline__ void blk_max_reduce(float* red, int t, float v, unsigned* gdst) {
    red[t] = v;
    __syncthreads();
    for (int s = 128; s > 0; s >>= 1) {
        if (t < s) red[t] = fmaxf(red[t], red[t + s]);
        __syncthreads();
    }
    if (t == 0) atomicMax(gdst, __float_as_uint(red[0]));
}

__device__ __forceinline__ int quant8(float v, float inv_s) {
    float q = rintf(v * inv_s);
    q = fminf(fmaxf(q, -128.0f), 127.0f);
    return (int)q;
}

__global__ void init_scalars(unsigned* scal) {
    if (threadIdx.x < 4) scal[threadIdx.x] = 0u;
}

__global__ void quant_weights(const float* __restrict__ w1,
                              const float* __restrict__ w2,
                              float* __restrict__ w1q,
                              float* __restrict__ ws2,
                              signed char* __restrict__ wq8) {
    __shared__ float red[256];
    const int blk = blockIdx.x, t = threadIdx.x;
    if (blk < DIM) {
        const int c = blk;
        float v = 0.0f;
        if (t < 81) v = fabsf(w1[c * 81 + t]);
        red[t] = v;
        __syncthreads();
        for (int s = 128; s > 0; s >>= 1) { if (t < s) red[t] = fmaxf(red[t], red[t + s]); __syncthreads(); }
        const float ws  = red[0] / 127.0f;
        const float inv = (ws > 0.0f) ? 1.0f / ws : 0.0f;
        if (t < 81) {
            w1q[c * 81 + t] = (float)quant8(w1[c * 81 + t], inv) * ws;
        }
    } else {
        const int o = blk - DIM;
        float v = 0.0f;
        for (int e = 0; e < 3; e++) v = fmaxf(v, fabsf(w2[o * DIM + t * 3 + e]));
        red[t] = v;
        __syncthreads();
        for (int s = 128; s > 0; s >>= 1) { if (t < s) red[t] = fmaxf(red[t], red[t + s]); __syncthreads(); }
        const float ws  = red[0] / 127.0f;
        const float inv = (ws > 0.0f) ? 1.0f / ws : 0.0f;
        if (t == 0) ws2[o] = ws;
        for (int e = 0; e < 3; e++) {
            const int i = t * 3 + e;
            wq8[o * DIM + i] = (signed char)quant8(w2[o * DIM + i], inv);
        }
    }
}

__global__ void __launch_bounds__(256) dwconv_gelu(const float* __restrict__ x,
                                                   const float* __restrict__ w1q,
                                                   const float* __restrict__ b1,
                                                   float* __restrict__ g,
                                                   unsigned* __restrict__ maxg) {
    __shared__ float xs[1024];
    __shared__ float wk[81];
    __shared__ float red[256];
    const int plane = blockIdx.x;
    const int c = plane % DIM;
    const float* xp = x + (size_t)plane * HW;
    const int t = threadIdx.x;
    for (int e = 0; e < 4; e++) xs[t + e * 256] = xp[t + e * 256];
    if (t < 81) wk[t] = w1q[c * 81 + t];
    const float bias = b1[c];
    __syncthreads();
    float lmax = 0.0f;
    for (int e = 0; e < 4; e++) {
        const int idx = t + e * 256;
        const int oy = idx >> 5, ox = idx & 31;
        float acc = bias;
        #pragma unroll
        for (int ky = 0; ky < 9; ky++) {
            const int iy = oy + ky - 4;
            if (iy < 0 || iy > 31) continue;
            #pragma unroll
            for (int kx = 0; kx < 9; kx++) {
                const int ix = ox + kx - 4;
                if (ix < 0 || ix > 31) continue;
                acc += xs[iy * 32 + ix] * wk[ky * 9 + kx];
            }
        }
        const float h = gelu_exact(acc);
        lmax = fmaxf(lmax, fabsf(h));
        g[(size_t)plane * HW + idx] = h;
    }
    blk_max_reduce(red, t, lmax, maxg);
}

__global__ void __launch_bounds__(256) quant_bn_add(const float* __restrict__ gin,
                                                    const float* __restrict__ x,
                                                    const float* __restrict__ g1,
                                                    const float* __restrict__ beta1,
                                                    const float* __restrict__ m1,
                                                    const float* __restrict__ v1,
                                                    float* __restrict__ yout,
                                                    const unsigned* __restrict__ maxg,
                                                    unsigned* __restrict__ maxy) {
    __shared__ float red[256];
    const float sg    = __uint_as_float(*maxg) / 127.0f;
    const float invsg = (sg > 0.0f) ? 1.0f / sg : 0.0f;
    const int t = threadIdx.x;
    float lmax = 0.0f;
    for (int e = 0; e < 8; e++) {
        const size_t idx = (size_t)blockIdx.x * 2048 + (size_t)e * 256 + t;
        const int c = (int)((idx >> 10) % DIM);
        const float gq  = (float)quant8(gin[idx], invsg) * sg;
        const float inv = g1[c] * rsqrtf(v1[c] + 1e-5f);
        const float yv  = gq * inv + (beta1[c] - m1[c] * inv) + x[idx];
        yout[idx] = yv;
        lmax = fmaxf(lmax, fabsf(yv));
    }
    blk_max_reduce(red, t, lmax, maxy);
}

__global__ void __launch_bounds__(256) pw_gemm_iu8(const float* __restrict__ y,
                                                   const signed char* __restrict__ wq8,
                                                   const float* __restrict__ ws2,
                                                   const float* __restrict__ b2,
                                                   float* __restrict__ hout,
                                                   const unsigned* __restrict__ maxy,
                                                   unsigned* __restrict__ maxh) {
    __shared__ __align__(16) signed char ldsA[128 * 64];
    __shared__ __align__(16) signed char ldsB[128 * 64];
    __shared__ float red[256];

    const int t     = threadIdx.x;
    const int n0    = blockIdx.x * 128;
    const int mg0   = blockIdx.y * 128;
    const int b_img = mg0 >> 10;
    const int hw0   = mg0 & (HW - 1);

    const float sy    = __uint_as_float(*maxy) / 127.0f;
    const float invsy = (sy > 0.0f) ? 1.0f / sy : 0.0f;

    const int lane   = t & 31;
    const int wv     = t >> 5;
    const int mstrip = wv * 16;
    const int l15    = lane & 15;
    const int lhi    = lane >> 4;

    v8i acc[8];
    #pragma unroll
    for (int i = 0; i < 8; i++) acc[i] = (v8i){0, 0, 0, 0, 0, 0, 0, 0};

    for (int kc = 0; kc < 12; kc++) {
        const int c0 = kc * 64;
        for (int e = 0; e < 8; e++) {
            const int idx = e * 256 + t;
            const int j   = idx & 127;
            const int cg  = idx >> 7;
            const size_t gbase = ((size_t)b_img * DIM + (c0 + cg * 4)) * HW + hw0 + j;
            const int q0 = quant8(y[gbase          ], invsy) & 255;
            const int q1 = quant8(y[gbase + 1 * HW ], invsy) & 255;
            const int q2 = quant8(y[gbase + 2 * HW ], invsy) & 255;
            const int q3 = quant8(y[gbase + 3 * HW ], invsy) & 255;
            *(int*)(ldsA + j * 64 + cg * 4) = q0 | (q1 << 8) | (q2 << 16) | (q3 << 24);
        }
        {
            const int n   = t >> 1;
            const int seg = (t & 1) * 32;
            const signed char* src = wq8 + ((size_t)(n0 + n)) * DIM + c0 + seg;
            *(int4*)(ldsB + n * 64 + seg)      = *(const int4*)(src);
            *(int4*)(ldsB + n * 64 + seg + 16) = *(const int4*)(src + 16);
        }
        __syncthreads();

        const int m  = mstrip + l15;
        const int kb = lhi * 16;
        const int2* ap = (const int2*)(ldsA + m * 64 + kb);
        const int2 a0 = ap[0], a1 = ap[1], a2 = ap[4], a3 = ap[5];
        v8i a;
        a[0] = a0.x; a[1] = a0.y; a[2] = a1.x; a[3] = a1.y;
        a[4] = a2.x; a[5] = a2.y; a[6] = a3.x; a[7] = a3.y;

        #pragma unroll
        for (int ns = 0; ns < 8; ns++) {
            const int n   = ns * 16 + l15;
            const int kb2 = lhi * 16;
            const int4* bp = (const int4*)(ldsB + n * 64 + kb2);
            const int4 b0 = bp[0], b1 = bp[2];
            v8i bb;
            bb[0] = b0.x; bb[1] = b0.y; bb[2] = b0.z; bb[3] = b0.w;
            bb[4] = b1.x; bb[5] = b1.y; bb[6] = b1.z; bb[7] = b1.w;
            acc[ns] = __builtin_amdgcn_wmma_i32_16x16x64_iu8(
                true, a, true, bb, acc[ns], false, false);
        }
        __syncthreads();
    }

    float lmax = 0.0f;
    #pragma unroll
    for (int ns = 0; ns < 8; ns++) {
        const int n     = n0 + ns * 16 + l15;
        const float scl = sy * ws2[n];
        const float bia = b2[n];
        #pragma unroll
        for (int r = 0; r < 8; r++) {
            const int ml  = mstrip + r + (lhi << 3);
            const float z = scl * (float)acc[ns][r] + bia;
            const float h = gelu_exact(z);
            lmax = fmaxf(lmax, fabsf(h));
            hout[((size_t)b_img * DIM + n) * HW + hw0 + ml] = h;
        }
    }
    blk_max_reduce(red, t, lmax, maxh);
}

__global__ void __launch_bounds__(256) quant_bn2(float* __restrict__ hio,
                                                 const float* __restrict__ g2,
                                                 const float* __restrict__ beta2,
                                                 const float* __restrict__ m2,
                                                 const float* __restrict__ v2,
                                                 const unsigned* __restrict__ maxh,
                                                 unsigned* __restrict__ maxo) {
    __shared__ float red[256];
    const float sh    = __uint_as_float(*maxh) / 127.0f;
    const float invsh = (sh > 0.0f) ? 1.0f / sh : 0.0f;
    const int t = threadIdx.x;
    float lmax = 0.0f;
    for (int e = 0; e < 8; e++) {
        const size_t idx = (size_t)blockIdx.x * 2048 + (size_t)e * 256 + t;
        const int c = (int)((idx >> 10) % DIM);
        const float hq  = (float)quant8(hio[idx], invsh) * sh;
        const float inv = g2[c] * rsqrtf(v2[c] + 1e-5f);
        const float ov  = hq * inv + (beta2[c] - m2[c] * inv);
        hio[idx] = ov;
        lmax = fmaxf(lmax, fabsf(ov));
    }
    blk_max_reduce(red, t, lmax, maxo);
}

__global__ void __launch_bounds__(256) final_quant(float* __restrict__ out,
                                                   const unsigned* __restrict__ maxo,
                                                   int total) {
    const float so    = __uint_as_float(*maxo) / 127.0f;
    const float invso = (so > 0.0f) ? 1.0f / so : 0.0f;
    for (int e = 0; e < 4; e++) {
        const size_t idx = (size_t)blockIdx.x * 1024 + (size_t)e * 256 + threadIdx.x;
        out[idx] = (float)quant8(out[idx], invso) * so;
    }
    if (blockIdx.x == 0 && threadIdx.x == 0) out[total] = so;
}

extern "C" void kernel_launch(void* const* d_in, const int* in_sizes, int n_in,
                              void* d_out, int out_size, void* d_ws, size_t ws_size,
                              hipStream_t stream) {
    const float* x     = (const float*)d_in[0];
    const float* w1    = (const float*)d_in[2];
    const float* b1    = (const float*)d_in[3];
    const float* g1    = (const float*)d_in[4];
    const float* beta1 = (const float*)d_in[5];
    const float* m1    = (const float*)d_in[6];
    const float* v1    = (const float*)d_in[7];
    const float* w2    = (const float*)d_in[8];
    const float* b2    = (const float*)d_in[9];
    const float* g2    = (const float*)d_in[10];
    const float* beta2 = (const float*)d_in[11];
    const float* m2    = (const float*)d_in[12];
    const float* v2    = (const float*)d_in[13];

    unsigned char* ws = (unsigned char*)d_ws;
    unsigned*    scal = (unsigned*)ws;
    float*       ws2  = (float*)(ws + 256);
    float*       w1q  = (float*)(ws + 4096);
    signed char* wq8  = (signed char*)(ws + 262144);
    float*       buf0 = (float*)(ws + (1u << 20));
    float*       bufh = (float*)d_out;

    init_scalars<<<1, 32, 0, stream>>>(scal);
    quant_weights<<<2 * DIM, 256, 0, stream>>>(w1, w2, w1q, ws2, wq8);
    dwconv_gelu<<<BATCH * DIM, 256, 0, stream>>>(x, w1q, b1, buf0, scal + 0);
    quant_bn_add<<<TOTAL / 2048, 256, 0, stream>>>(buf0, x, g1, beta1, m1, v1,
                                                   buf0, scal + 0, scal + 1);
    pw_gemm_iu8<<<dim3(DIM / 128, (BATCH * HW) / 128), 256, 0, stream>>>(
        buf0, wq8, ws2, b2, bufh, scal + 1, scal + 2);
    quant_bn2<<<TOTAL / 2048, 256, 0, stream>>>(bufh, g2, beta2, m2, v2,
                                                scal + 2, scal + 3);
    final_quant<<<TOTAL / 1024, 256, 0, stream>>>((float*)d_out, scal + 3, TOTAL);
}
